// VoxelSampler_anchor_6657199309193
// MI455X (gfx1250) — 8-bit matrix operand order repaired and checked by reading, compiled, not run on hardware
//
#include <hip/hip_runtime.h>
#include <hip/hip_bf16.h>

#define VOXEL     0.4f
#define PCX      -75.2f
#define PCY      -75.2f
#define IOU_TH    0.5f
#define MMAXBOX   128
#define NSAMPLE   256
#define NANCH     3
#define TPB       256
#define KSPLIT    16

typedef int v8i __attribute__((ext_vector_type(8)));

__device__ __forceinline__ size_t a_frag_addr(int m, int k, int nkb) {
  int mt = m >> 4, mr = m & 15;
  int kb = k >> 6, kr = k & 63;
  int half = (kr >> 3) & 1;
  int lane = mr + (half << 4);
  int kk   = kr & 7;
  int g16  = kr >> 4;
  int v    = (g16 << 1) + (kk >> 2);
  int off  = (v << 2) + (kk & 3);
  return (((size_t)mt * (size_t)nkb + (size_t)kb) << 10) + (size_t)(lane << 5) + (size_t)off;
}

__device__ __forceinline__ size_t b_frag_addr(int j, int k, int nkb) {
  int jt = j >> 4, jr = j & 15;
  int kb = k >> 6, kr = k & 63;
  int h32 = kr >> 4;
  int w   = kr & 7;
  int lane = jr + (((kr >> 3) & 1) << 4);
  int kk   = w;
  int v    = (h32 << 1) + (kk >> 2);
  int off  = (v << 2) + (kk & 3);
  return (((size_t)jt * (size_t)nkb + (size_t)kb) << 10) + (size_t)(lane << 5) + (size_t)off;
}

__global__ void k_vmask(const float* __restrict__ points, const float* __restrict__ boxes,
                        unsigned char* __restrict__ vmask, int N, int M) {
  __shared__ float sqx[MMAXBOX], sqy[MMAXBOX], srv[MMAXBOX];
  for (int m = threadIdx.x; m < M; m += blockDim.x) {
    float bx = boxes[m * 7 + 0], by = boxes[m * 7 + 1];
    float dx = boxes[m * 7 + 3], dy = boxes[m * 7 + 4];
    sqx[m] = floorf((bx - PCX) / VOXEL);
    sqy[m] = floorf((by - PCY) / VOXEL);
    srv[m] = ceilf(0.5f * sqrtf(dx * dx + dy * dy) / VOXEL);
  }
  __syncthreads();
  int n = blockIdx.x * blockDim.x + threadIdx.x;
  if (n >= N) return;
  float cx = floorf((points[(size_t)n * 5 + 0] - PCX) / VOXEL);
  float cy = floorf((points[(size_t)n * 5 + 1] - PCY) / VOXEL);
  unsigned char v = 0;
  for (int m = 0; m < M; ++m) {
    if (fabsf(sqx[m] - cx) < srv[m] && fabsf(sqy[m] - cy) < srv[m]) { v = 1; break; }
  }
  vmask[n] = v;
}

__global__ void k_pmf(const float* __restrict__ points, const float* __restrict__ boxes,
                      const unsigned char* __restrict__ vmask,
                      unsigned char* __restrict__ afrag, unsigned char* __restrict__ bfrag,
                      int N, int nkb) {
  int m = blockIdx.y;
  int k0 = (blockIdx.x * blockDim.x + threadIdx.x) << 2;
  int kpad = nkb << 6;
  if (k0 >= kpad) return;
  float bx = boxes[m * 7 + 0], by = boxes[m * 7 + 1];
  float ddx = boxes[m * 7 + 3], ddy = boxes[m * 7 + 4];
  float r2 = 0.25f * (ddx * ddx + ddy * ddy);
  unsigned int pack = 0;
  for (int i = 0; i < 4; ++i) {
    int k = k0 + i;
    unsigned int v = 0;
    if (k < N) {
      float px = points[(size_t)k * 5 + 0] - bx;
      float py = points[(size_t)k * 5 + 1] - by;
      v = (px * px + py * py <= r2 && vmask[k]) ? 1u : 0u;
    }
    pack |= v << (i * 8);
  }
  *(unsigned int*)(afrag + a_frag_addr(m, k0, nkb)) = pack;
  *(unsigned int*)(bfrag + b_frag_addr(m, k0, nkb)) = pack;
}

__global__ void __launch_bounds__(32) k_gemm(const unsigned char* __restrict__ afrag,
                                             const unsigned char* __restrict__ bfrag,
                                             int* __restrict__ partials, int nkb, int M) {
  int lane = threadIdx.x;
  int half = (M >> 4) >> 1;
  int bi = (blockIdx.x / half) * 2;
  int bj = (blockIdx.x % half) * 2;
  int ks = blockIdx.y;
  int chunk = (nkb + KSPLIT - 1) / KSPLIT;
  int kb0 = ks * chunk;
  int kb1 = kb0 + chunk; if (kb1 > nkb) kb1 = nkb;

  v8i c00{}, c01{}, c10{}, c11{};
  const v8i* A0 = (const v8i*)(afrag + (((size_t)(bi    ) * nkb) << 10));
  const v8i* A1 = (const v8i*)(afrag + (((size_t)(bi + 1) * nkb) << 10));
  const v8i* B0 = (const v8i*)(bfrag + (((size_t)(bj    ) * nkb) << 10));
  const v8i* B1 = (const v8i*)(bfrag + (((size_t)(bj + 1) * nkb) << 10));
  for (int kb = kb0; kb < kb1; ++kb) {
    size_t idx = (size_t)kb * 32 + lane;
    v8i a0 = A0[idx], a1 = A1[idx];
    v8i b0 = B0[idx], b1 = B1[idx];
    c00 = __builtin_amdgcn_wmma_i32_16x16x64_iu8(false, a0, false, b0, c00, false, false);
    c01 = __builtin_amdgcn_wmma_i32_16x16x64_iu8(false, a0, false, b1, c01, false, false);
    c10 = __builtin_amdgcn_wmma_i32_16x16x64_iu8(false, a1, false, b0, c10, false, false);
    c11 = __builtin_amdgcn_wmma_i32_16x16x64_iu8(false, a1, false, b1, c11, false, false);
  }
  int* part = partials + (size_t)ks * M * M;
  int rowh = (lane >> 4) << 3;
  int col  = lane & 15;
  for (int r = 0; r < 8; ++r) {
    int m0 = bi * 16 + r + rowh, m1 = (bi + 1) * 16 + r + rowh;
    int n0 = bj * 16 + col,      n1 = (bj + 1) * 16 + col;
    part[m0 * M + n0] = c00[r];
    part[m0 * M + n1] = c01[r];
    part[m1 * M + n0] = c10[r];
    part[m1 * M + n1] = c11[r];
  }
}

__global__ void k_reduce(const int* __restrict__ partials, int* __restrict__ inter, int MM) {
  int idx = blockIdx.x * blockDim.x + threadIdx.x;
  if (idx >= MM) return;
  int s = 0;
  for (int p = 0; p < KSPLIT; ++p) s += partials[(size_t)p * MM + idx];
  inter[idx] = s;
}

__global__ void k_group(const int* __restrict__ inter, const int* __restrict__ labels,
                        int* __restrict__ wsAnchors, int* __restrict__ outAnchors, int M) {
  __shared__ float mmax[MMAXBOX];
  __shared__ int   midx[MMAXBOX];
  __shared__ int   aidx[MMAXBOX * NANCH];
  __shared__ int   gof[MMAXBOX];
  __shared__ int   gcnt[MMAXBOX];
  int i = threadIdx.x;
  if (i < M) {
    int li = labels[i];
    int cnti = inter[i * M + i];
    float best = -1.0f; int bidx = 0;
    for (int j = 0; j < M; ++j) {
      float val = 0.0f;
      if (j != i && labels[j] == li) {
        int in = inter[i * M + j];
        float uni = (float)(cnti + inter[j * M + j] - in);
        if (uni > 0.0f) val = (float)in / fmaxf(uni, 1.0f);
      }
      if (val > best) { best = val; bidx = j; }
    }
    mmax[i] = best; midx[i] = bidx;
    gof[i] = -1; gcnt[i] = 0;
    aidx[i * 3 + 0] = aidx[i * 3 + 1] = aidx[i * 3 + 2] = -1;
  }
  __syncthreads();
  if (i == 0) {
    int ng = 0;
    for (int t = 0; t < M; ++t) {
      int j = midx[t];
      int gj = gof[j];
      int gjs = gj > 0 ? gj : 0;
      bool join = (mmax[t] > IOU_TH) && (gj >= 0) && (gcnt[gjs] < NANCH);
      int tg = join ? gjs : ng;
      int ts = join ? gcnt[gjs] : 0;
      aidx[tg * 3 + ts] = t;
      gof[t] = tg;
      gcnt[tg] += 1;
      if (!join) ++ng;
    }
  }
  __syncthreads();
  if (i < M) {
    for (int a = 0; a < NANCH; ++a) {
      wsAnchors[i * 3 + a]  = aidx[i * 3 + a];
      outAnchors[i * 3 + a] = aidx[i * 3 + a];
    }
  }
}

__global__ void __launch_bounds__(TPB) k_sample(const float* __restrict__ points,
                                                const float* __restrict__ boxes,
                                                const unsigned char* __restrict__ vmask,
                                                const int* __restrict__ wsAnchors,
                                                float* __restrict__ out, int N) {
  int g = blockIdx.x, tid = threadIdx.x;
  float* og = out + (size_t)g * NSAMPLE * 5;
  for (int t = tid; t < NSAMPLE * 5; t += TPB) og[t] = 0.0f;

  __shared__ float mcx[NANCH], mcy[NANCH], mr2[NANCH];
  __shared__ int nmem, cnt[4], run[4], wtot[TPB / 32][4], donef;
  if (tid == 0) {
    int nm = 0;
    for (int a = 0; a < NANCH; ++a) {
      int bi = wsAnchors[g * 3 + a];
      if (bi >= 0) {
        float ddx = boxes[bi * 7 + 3], ddy = boxes[bi * 7 + 4];
        mcx[nm] = boxes[bi * 7 + 0]; mcy[nm] = boxes[bi * 7 + 1];
        mr2[nm] = 0.25f * (ddx * ddx + ddy * ddy);
        ++nm;
      }
    }
    nmem = nm;
    cnt[0] = cnt[1] = cnt[2] = cnt[3] = 0;
    run[1] = run[2] = run[3] = 0;
    donef = 0;
  }
  __syncthreads();
  if (nmem == 0) return;

  int wave = tid >> 5, lane = tid & 31;
  int nchunks = (N + TPB - 1) / TPB;

  for (int kb = 0; kb < nchunks; ++kb) {
    int k = kb * TPB + tid;
    int v = 0;
    if (k < N && vmask[k]) {
      float px = points[(size_t)k * 5 + 0], py = points[(size_t)k * 5 + 1];
      for (int a = 0; a < nmem; ++a) {
        float dx = px - mcx[a], dy = py - mcy[a];
        if (dx * dx + dy * dy <= mr2[a]) ++v;
      }
    }
    if (v) atomicAdd(&cnt[v], 1);
  }
  __syncthreads();
  int base[4];
  base[0] = 0; base[3] = 0; base[2] = cnt[3]; base[1] = cnt[3] + cnt[2];

  for (int kb = 0; kb < nchunks; ++kb) {
    int k = kb * TPB + tid;
    int v = 0;
    if (k < N && vmask[k]) {
      float px = points[(size_t)k * 5 + 0], py = points[(size_t)k * 5 + 1];
      for (int a = 0; a < nmem; ++a) {
        float dx = px - mcx[a], dy = py - mcy[a];
        if (dx * dx + dy * dy <= mr2[a]) ++v;
      }
    }
    unsigned int lowmask = (lane == 0) ? 0u : (0xFFFFFFFFu >> (32 - lane));
    int pre = 0;
    for (int c = 1; c <= 3; ++c) {
      unsigned int b32 = (unsigned int)__ballot(v == c);
      if (lane == 0) wtot[wave][c] = __popc(b32);
      if (v == c) pre = __popc(b32 & lowmask);
    }
    __syncthreads();
    if (v > 0) {
      int cross = 0;
      for (int w = 0; w < wave; ++w) cross += wtot[w][v];
      int slot = base[v] + run[v] + cross + pre;
      if (slot < NSAMPLE) {
        float* o = og + slot * 5;
        const float* p = points + (size_t)k * 5;
        o[0] = p[0]; o[1] = p[1]; o[2] = p[2]; o[3] = p[3]; o[4] = p[4];
      }
    }
    __syncthreads();
    if (tid == 0) {
      bool d = true;
      for (int c = 1; c <= 3; ++c) {
        int tot = 0;
        for (int w = 0; w < TPB / 32; ++w) tot += wtot[w][c];
        run[c] += tot;
        if (run[c] < cnt[c] && base[c] + run[c] < NSAMPLE) d = false;
      }
      donef = d ? 1 : 0;
    }
    __syncthreads();
    if (donef) break;
  }
}

extern "C" void kernel_launch(void* const* d_in, const int* in_sizes, int n_in,
                              void* d_out, int out_size, void* d_ws, size_t ws_size,
                              hipStream_t stream) {
  const float* points = (const float*)d_in[0];
  const float* boxes  = (const float*)d_in[1];
  const int*   labels = (const int*)d_in[2];
  int N = in_sizes[0] / 5;
  int M = in_sizes[1] / 7;
  if (M > MMAXBOX) M = MMAXBOX;

  int nkb = (N + 63) / 64;
  size_t fragBytes = (size_t)((M + 15) / 16) * (size_t)nkb * 1024;
  unsigned char* ws    = (unsigned char*)d_ws;
  unsigned char* afrag = ws;
  unsigned char* bfrag = ws + fragBytes;
  unsigned char* vmask = bfrag + fragBytes;
  size_t vmaskBytes = (((size_t)nkb * 64) + 1023) & ~(size_t)1023;
  int* partials  = (int*)(vmask + vmaskBytes);
  int* inter     = partials + (size_t)KSPLIT * M * M;
  int* wsAnchors = inter + (size_t)M * M;
  float* out        = (float*)d_out;
  int*   outAnchors = (int*)(out + (size_t)M * NSAMPLE * 5);

  k_vmask<<<(N + 255) / 256, 256, 0, stream>>>(points, boxes, vmask, N, M);

  int kq = nkb * 16;
  dim3 g2((unsigned)((kq + 255) / 256), (unsigned)M);
  k_pmf<<<g2, 256, 0, stream>>>(points, boxes, vmask, afrag, bfrag, N, nkb);

  int half = (M / 16) / 2;
  dim3 g3((unsigned)(half * half), KSPLIT);
  k_gemm<<<g3, 32, 0, stream>>>(afrag, bfrag, partials, nkb, M);

  int MM = M * M;
  k_reduce<<<(MM + 255) / 256, 256, 0, stream>>>(partials, inter, MM);

  k_group<<<1, M, 0, stream>>>(inter, labels, wsAnchors, outAnchors, M);

  k_sample<<<M, TPB, 0, stream>>>(points, boxes, vmask, wsAnchors, out, N);
}
